// OptimizedAttentionBlock_59511066853632
// MI455X (gfx1250) — matrix operand order repaired and checked by reading, not compiled, not run on hardware
//
#include <hip/hip_runtime.h>
#include <hip/hip_bf16.h>

typedef __attribute__((ext_vector_type(16))) _Float16 v16h;
typedef __attribute__((ext_vector_type(8)))  float    v8f;

union FragH { v16h h; unsigned int u[8]; };
union FragF { v8f  f; float s[8]; };

#define NB    16
#define NC    512
#define NHW   1024
#define NHEAD 8
#define HD    64
#define NGRP  32
#define CPG   16   // NC/NGRP

#if defined(__gfx1250__) && __has_builtin(__builtin_amdgcn_global_load_async_to_lds_b128) && __has_builtin(__builtin_amdgcn_s_wait_asynccnt)
#define USE_ASYNC 1
#endif

#ifdef USE_ASYNC
typedef int i128 __attribute__((__vector_size__(16)));
typedef __attribute__((address_space(1))) i128* gp128;
typedef __attribute__((address_space(3))) i128* lp128;
#endif

// copy 16 bytes global -> LDS (async-to-LDS on gfx1250, register fallback otherwise)
__device__ __forceinline__ void cp16(const _Float16* g, _Float16* l) {
#ifdef USE_ASYNC
    __builtin_amdgcn_global_load_async_to_lds_b128(
        (gp128)(i128*)(void*)g, (lp128)(i128*)l, 0, 0);
#else
    *(uint4*)l = *(const uint4*)g;
#endif
}

__device__ __forceinline__ void async_fence() {
#ifdef USE_ASYNC
    __builtin_amdgcn_s_wait_asynccnt(0);
#endif
}

// ---------------------------------------------------------------- gn stats
__global__ void gn_stats_k(const float* __restrict__ x, float* __restrict__ mean,
                           float* __restrict__ rstd) {
    int bg = blockIdx.x;                           // b*32 + g
    const float* p = x + (size_t)bg * (CPG * NHW); // contiguous group slab
    float s = 0.f, s2 = 0.f;
    for (int i = threadIdx.x; i < CPG * NHW; i += 256) {
        float v = p[i]; s += v; s2 += v * v;
    }
    __shared__ float sh1[256], sh2[256];
    sh1[threadIdx.x] = s; sh2[threadIdx.x] = s2;
    __syncthreads();
    for (int st = 128; st > 0; st >>= 1) {
        if (threadIdx.x < st) {
            sh1[threadIdx.x] += sh1[threadIdx.x + st];
            sh2[threadIdx.x] += sh2[threadIdx.x + st];
        }
        __syncthreads();
    }
    if (threadIdx.x == 0) {
        float m   = sh1[0] * (1.f / (CPG * NHW));
        float var = sh2[0] * (1.f / (CPG * NHW)) - m * m;
        mean[bg] = m;
        rstd[bg] = rsqrtf(var + 1e-5f);
    }
}

// ---------------------------------------------------------------- weight cvt
__global__ void wcvt_k(const float* __restrict__ qkvw, const float* __restrict__ projw,
                       _Float16* __restrict__ qh, _Float16* __restrict__ ph) {
    int i = blockIdx.x * 256 + threadIdx.x;
    if (i < 3 * NC * NC) qh[i] = (_Float16)qkvw[i];
    if (i < NC * NC)     ph[i] = (_Float16)projw[i];
}

// ---------------------------------------------------------------- gn apply + transpose
__global__ void gn_apply_k(const float* __restrict__ x, const float* __restrict__ gw,
                           const float* __restrict__ gb, const float* __restrict__ mean,
                           const float* __restrict__ rstd, float* __restrict__ xn,
                           _Float16* __restrict__ xnT) {
    __shared__ _Float16 tile[32][33];
    int b = blockIdx.z, ct = blockIdx.y, ht = blockIdx.x;
    int tx = threadIdx.x, ty = threadIdx.y;
#pragma unroll
    for (int i = 0; i < 4; ++i) {
        int c  = ct * 32 + ty + i * 8;
        int hw = ht * 32 + tx;
        int g  = c >> 4;
        size_t idx = ((size_t)b * NC + c) * NHW + hw;
        float v = (x[idx] - mean[b * NGRP + g]) * rstd[b * NGRP + g] * gw[c] + gb[c];
        xn[idx] = v;                       // fp32 copy kept for the residual
        tile[ty + i * 8][tx] = (_Float16)v;
    }
    __syncthreads();
#pragma unroll
    for (int i = 0; i < 4; ++i) {
        int hwl = ty + i * 8;
        int cl  = tx;
        xnT[((size_t)b * NHW + ht * 32 + hwl) * NC + ct * 32 + cl] = tile[cl][hwl];
    }
}

// ---------------------------------------------------------------- QKV GEMM (WMMA)
// A = xnT [16384 x 512] f16 ; B^T = qkv_w f16 [o, c] ; WG tile 64x128, wave tile 32x64
__global__ __launch_bounds__(128) void qkv_gemm_k(const _Float16* __restrict__ xnT,
                                                  const _Float16* __restrict__ wh,
                                                  _Float16* __restrict__ qo,
                                                  _Float16* __restrict__ ko,
                                                  _Float16* __restrict__ vto) {
    __shared__ __attribute__((aligned(16))) _Float16 As[2][64 * 32];
    __shared__ __attribute__((aligned(16))) _Float16 Bs[2][128 * 32];
    __shared__ __attribute__((aligned(16))) _Float16 Ot[64 * 130];
    const int t = threadIdx.x, lane = t & 31, w = t >> 5;
    const int lm = lane & 15, hi = lane >> 4;
    const int Mbase = blockIdx.x * 64;
    const int Nbase = blockIdx.y * 128;
    const int wm = (w >> 1) * 32, wn = (w & 1) * 64;

    FragF acc[2][4];
#pragma unroll
    for (int i = 0; i < 2; ++i)
#pragma unroll
        for (int n = 0; n < 4; ++n)
#pragma unroll
            for (int r = 0; r < 8; ++r) acc[i][n].s[r] = 0.f;

    // per-thread fixed (row, chunk) assignments, global bases hoisted out of K-loop
    const int arow = t >> 1, ac8 = (t & 1) * 16;            // A: 64 rows x 2 x 16 halves
    const int brow = t >> 1, bc8 = (t & 1) * 16;            // B: rows t/2 and 64+t/2
    const _Float16* gA = xnT + (size_t)(Mbase + arow) * NC + ac8;
    const _Float16* gB0 = wh + (size_t)(Nbase + brow) * NC + bc8;
    const _Float16* gB1 = wh + (size_t)(Nbase + 64 + brow) * NC + bc8;

    auto tile_load = [&](int buf, int kk) {
        int ko_ = kk * 32;
        cp16(gA + ko_,     &As[buf][arow * 32 + ac8]);
        cp16(gA + ko_ + 8, &As[buf][arow * 32 + ac8 + 8]);
        cp16(gB0 + ko_,     &Bs[buf][brow * 32 + bc8]);
        cp16(gB0 + ko_ + 8, &Bs[buf][brow * 32 + bc8 + 8]);
        cp16(gB1 + ko_,     &Bs[buf][(64 + brow) * 32 + bc8]);
        cp16(gB1 + ko_ + 8, &Bs[buf][(64 + brow) * 32 + bc8 + 8]);
    };

    tile_load(0, 0);
    async_fence();
    __syncthreads();
    for (int kk = 0; kk < 16; ++kk) {
        int cur = kk & 1;
        if (kk + 1 < 16) tile_load(cur ^ 1, kk + 1);
        FragH a[2], bfr[4];
#pragma unroll
        for (int j = 0; j < 8; ++j) {
            int ka = 2 * j + (j >= 4 ? 8 : 0) + hi * 8;   // A-frag K index
            int kb = 2 * j + (j >= 4 ? 8 : 0) + hi * 8;   // B-frag K index
            a[0].u[j] = *(const unsigned int*)&As[cur][(wm + lm) * 32 + ka];
            a[1].u[j] = *(const unsigned int*)&As[cur][(wm + 16 + lm) * 32 + ka];
#pragma unroll
            for (int n = 0; n < 4; ++n)
                bfr[n].u[j] = *(const unsigned int*)&Bs[cur][(wn + n * 16 + lm) * 32 + kb];
        }
#pragma unroll
        for (int i = 0; i < 2; ++i)
#pragma unroll
            for (int n = 0; n < 4; ++n)
                acc[i][n].f = __builtin_amdgcn_wmma_f32_16x16x32_f16(
                    false, a[i].h, false, bfr[n].h, (short)0, acc[i][n].f, false, false);
        async_fence();
        __syncthreads();
    }
    // stage D tiles (f16) for coalesced stores
#pragma unroll
    for (int i = 0; i < 2; ++i)
#pragma unroll
        for (int n = 0; n < 4; ++n)
#pragma unroll
            for (int r = 0; r < 8; ++r) {
                int row = wm + i * 16 + r + hi * 8;
                int col = wn + n * 16 + lm;
                Ot[row * 130 + col] = (_Float16)acc[i][n].s[r];
            }
    __syncthreads();
    int b = Mbase >> 10, hwb = Mbase & 1023;
    int which = Nbase / NC, headbase = (Nbase % NC) / HD;
    if (which < 2) {
        _Float16* dst = (which == 0) ? qo : ko;
#pragma unroll
        for (int it = 0; it < 64; ++it) {
            int i = t + it * 128;
            int row = i >> 7, col = i & 127;
            int head = headbase + (col >> 6), d = col & 63;
            size_t bh = (size_t)(b * NHEAD + head) * (NHW * HD);
            dst[bh + (size_t)(hwb + row) * HD + d] = Ot[row * 130 + col];
        }
    } else {                                       // V stored transposed [bh, d, hw]
#pragma unroll
        for (int it = 0; it < 64; ++it) {
            int i = t + it * 128;
            int col = i >> 6, r = i & 63;          // col = local o, r = local hw
            int head = headbase + (col >> 6), d = col & 63;
            size_t bh = (size_t)(b * NHEAD + head) * (NHW * HD);
            vto[bh + (size_t)d * NHW + hwb + r] = Ot[r * 130 + col];
        }
    }
}

// ---------------------------------------------------------------- flash attention
__global__ __launch_bounds__(128) void attn_k(const _Float16* __restrict__ q,
                                              const _Float16* __restrict__ kg,
                                              const _Float16* __restrict__ vt,
                                              _Float16* __restrict__ hbuf) {
    __shared__ __attribute__((aligned(16))) _Float16 Ks[2][64 * 64];   // [m][d]
    __shared__ __attribute__((aligned(16))) _Float16 Vs[2][64 * 64];   // [d][m]
    __shared__ __attribute__((aligned(16))) _Float16 Ps[4][16 * 64];
    const int t = threadIdx.x, lane = t & 31, w = t >> 5;
    const int lm = lane & 15, hi = lane >> 4;
    const int bh = blockIdx.y;
    const int b = bh >> 3, head = bh & 7;
    const int qb = blockIdx.x * 64;
    const size_t base = (size_t)bh * (NHW * HD);

    FragH aq[2];                                   // Q fragments, resident
    const int qrow = qb + w * 16 + lm;
#pragma unroll
    for (int s = 0; s < 2; ++s)
#pragma unroll
        for (int j = 0; j < 8; ++j) {
            int kd = 2 * j + (j >= 4 ? 8 : 0) + hi * 8 + s * 32;
            aq[s].u[j] = *(const unsigned int*)&q[base + (size_t)qrow * HD + kd];
        }

    float mrow[8], lrow[8];
    FragF ho[4];
#pragma unroll
    for (int r = 0; r < 8; ++r) { mrow[r] = -1e30f; lrow[r] = 0.f; }
#pragma unroll
    for (int ot = 0; ot < 4; ++ot)
#pragma unroll
        for (int r = 0; r < 8; ++r) ho[ot].s[r] = 0.f;

    // per-thread fixed copy slots: 16 rows per wave-quarter, 8x16B per row pair
    const int krow = t >> 1, kc8 = (t & 1) * 16;   // rows t/2 and 32+t/2, 2 chunks each
    const _Float16* gK0 = kg + base + (size_t)krow * HD + kc8;
    const _Float16* gK1 = kg + base + (size_t)(32 + krow) * HD + kc8;
    const _Float16* gV0 = vt + base + (size_t)krow * NHW + kc8;
    const _Float16* gV1 = vt + base + (size_t)(32 + krow) * NHW + kc8;

    auto chunk_load = [&](int buf, int c) {
        size_t kofs = (size_t)c * 64 * HD;         // K advances by 64 rows
        int    vofs = c * 64;                      // V advances by 64 columns
        cp16(gK0 + kofs,     &Ks[buf][krow * 64 + kc8]);
        cp16(gK0 + kofs + 8, &Ks[buf][krow * 64 + kc8 + 8]);
        cp16(gK1 + kofs,     &Ks[buf][(32 + krow) * 64 + kc8]);
        cp16(gK1 + kofs + 8, &Ks[buf][(32 + krow) * 64 + kc8 + 8]);
        cp16(gV0 + vofs,     &Vs[buf][krow * 64 + kc8]);
        cp16(gV0 + vofs + 8, &Vs[buf][krow * 64 + kc8 + 8]);
        cp16(gV1 + vofs,     &Vs[buf][(32 + krow) * 64 + kc8]);
        cp16(gV1 + vofs + 8, &Vs[buf][(32 + krow) * 64 + kc8 + 8]);
    };

    chunk_load(0, 0);
    async_fence();
    __syncthreads();
    for (int c = 0; c < 16; ++c) {                 // 16 key chunks of 64
        int cur = c & 1;
        if (c + 1 < 16) chunk_load(cur ^ 1, c + 1);

        FragF sA[4];                               // S = scale * Q K^T (16x64)
#pragma unroll
        for (int nt = 0; nt < 4; ++nt) {
#pragma unroll
            for (int r = 0; r < 8; ++r) sA[nt].s[r] = 0.f;
#pragma unroll
            for (int s = 0; s < 2; ++s) {
                FragH bk;
#pragma unroll
                for (int j = 0; j < 8; ++j) {
                    int kd = 2 * j + (j >= 4 ? 8 : 0) + hi * 8 + s * 32;
                    int n  = nt * 16 + lm;
                    bk.u[j] = *(const unsigned int*)&Ks[cur][n * 64 + kd];
                }
                sA[nt].f = __builtin_amdgcn_wmma_f32_16x16x32_f16(
                    false, aq[s].h, false, bk.h, (short)0, sA[nt].f, false, false);
            }
        }
        // online softmax update (rows split across lane halves)
        float fac[8];
#pragma unroll
        for (int r = 0; r < 8; ++r) {
            float cm = -1e30f;
#pragma unroll
            for (int nt = 0; nt < 4; ++nt) {
                float v = sA[nt].s[r] * 0.125f;    // 1/sqrt(64)
                sA[nt].s[r] = v;
                cm = fmaxf(cm, v);
            }
#pragma unroll
            for (int off = 1; off < 16; off <<= 1) cm = fmaxf(cm, __shfl_xor(cm, off, 32));
            float mnew = fmaxf(mrow[r], cm);
            float f = __expf(mrow[r] - mnew);
            float ps = 0.f;
#pragma unroll
            for (int nt = 0; nt < 4; ++nt) {
                float p = __expf(sA[nt].s[r] - mnew);
                ps += p;
                Ps[w][(r + hi * 8) * 64 + nt * 16 + lm] = (_Float16)p;
            }
#pragma unroll
            for (int off = 1; off < 16; off <<= 1) ps += __shfl_xor(ps, off, 32);
            lrow[r] = lrow[r] * f + ps;
            mrow[r] = mnew;
            fac[r] = f;
        }
#pragma unroll
        for (int ot = 0; ot < 4; ++ot)
#pragma unroll
            for (int r = 0; r < 8; ++r) ho[ot].s[r] *= fac[r];
        // O += P V
#pragma unroll
        for (int s = 0; s < 2; ++s) {
            FragH ap;
#pragma unroll
            for (int j = 0; j < 8; ++j) {
                int km = 2 * j + (j >= 4 ? 8 : 0) + hi * 8 + s * 32;
                ap.u[j] = *(const unsigned int*)&Ps[w][lm * 64 + km];
            }
#pragma unroll
            for (int ot = 0; ot < 4; ++ot) {
                FragH bv;
#pragma unroll
                for (int j = 0; j < 8; ++j) {
                    int km = 2 * j + (j >= 4 ? 8 : 0) + hi * 8 + s * 32;
                    int n  = ot * 16 + lm;
                    bv.u[j] = *(const unsigned int*)&Vs[cur][n * 64 + km];
                }
                ho[ot].f = __builtin_amdgcn_wmma_f32_16x16x32_f16(
                    false, ap.h, false, bv.h, (short)0, ho[ot].f, false, false);
            }
        }
        async_fence();
        __syncthreads();
    }
    // normalize + store h in [b, hw, c] layout (A-friendly for proj GEMM)
#pragma unroll
    for (int r = 0; r < 8; ++r) {
        float inv = 1.f / lrow[r];
        int row = r + hi * 8;
        int hw = qb + w * 16 + row;
#pragma unroll
        for (int ot = 0; ot < 4; ++ot) {
            int cc = head * HD + ot * 16 + lm;
            hbuf[((size_t)b * NHW + hw) * NC + cc] = (_Float16)(ho[ot].s[r] * inv);
        }
    }
}

// ---------------------------------------------------------------- proj GEMM + bias + residual
// WG tile 64x128, wave tile 32x64
__global__ __launch_bounds__(128) void proj_gemm_k(const _Float16* __restrict__ hbuf,
                                                   const _Float16* __restrict__ wh,
                                                   const float* __restrict__ pb,
                                                   const float* __restrict__ xn,
                                                   float* __restrict__ out) {
    __shared__ __attribute__((aligned(16))) _Float16 As[2][64 * 32];
    __shared__ __attribute__((aligned(16))) _Float16 Bs[2][128 * 32];
    __shared__ __attribute__((aligned(16))) float Of[64 * 130];
    const int t = threadIdx.x, lane = t & 31, w = t >> 5;
    const int lm = lane & 15, hi = lane >> 4;
    const int Mbase = blockIdx.x * 64;
    const int Nbase = blockIdx.y * 128;
    const int wm = (w >> 1) * 32, wn = (w & 1) * 64;

    FragF acc[2][4];
#pragma unroll
    for (int i = 0; i < 2; ++i)
#pragma unroll
        for (int n = 0; n < 4; ++n)
#pragma unroll
            for (int r = 0; r < 8; ++r) acc[i][n].s[r] = 0.f;

    const int arow = t >> 1, ac8 = (t & 1) * 16;
    const int brow = t >> 1, bc8 = (t & 1) * 16;
    const _Float16* gA = hbuf + (size_t)(Mbase + arow) * NC + ac8;
    const _Float16* gB0 = wh + (size_t)(Nbase + brow) * NC + bc8;
    const _Float16* gB1 = wh + (size_t)(Nbase + 64 + brow) * NC + bc8;

    auto tile_load = [&](int buf, int kk) {
        int ko_ = kk * 32;
        cp16(gA + ko_,     &As[buf][arow * 32 + ac8]);
        cp16(gA + ko_ + 8, &As[buf][arow * 32 + ac8 + 8]);
        cp16(gB0 + ko_,     &Bs[buf][brow * 32 + bc8]);
        cp16(gB0 + ko_ + 8, &Bs[buf][brow * 32 + bc8 + 8]);
        cp16(gB1 + ko_,     &Bs[buf][(64 + brow) * 32 + bc8]);
        cp16(gB1 + ko_ + 8, &Bs[buf][(64 + brow) * 32 + bc8 + 8]);
    };

    tile_load(0, 0);
    async_fence();
    __syncthreads();
    for (int kk = 0; kk < 16; ++kk) {
        int cur = kk & 1;
        if (kk + 1 < 16) tile_load(cur ^ 1, kk + 1);
        FragH a[2], bfr[4];
#pragma unroll
        for (int j = 0; j < 8; ++j) {
            int ka = 2 * j + (j >= 4 ? 8 : 0) + hi * 8;
            int kb = 2 * j + (j >= 4 ? 8 : 0) + hi * 8;
            a[0].u[j] = *(const unsigned int*)&As[cur][(wm + lm) * 32 + ka];
            a[1].u[j] = *(const unsigned int*)&As[cur][(wm + 16 + lm) * 32 + ka];
#pragma unroll
            for (int n = 0; n < 4; ++n)
                bfr[n].u[j] = *(const unsigned int*)&Bs[cur][(wn + n * 16 + lm) * 32 + kb];
        }
#pragma unroll
        for (int i = 0; i < 2; ++i)
#pragma unroll
            for (int n = 0; n < 4; ++n)
                acc[i][n].f = __builtin_amdgcn_wmma_f32_16x16x32_f16(
                    false, a[i].h, false, bfr[n].h, (short)0, acc[i][n].f, false, false);
        async_fence();
        __syncthreads();
    }
#pragma unroll
    for (int i = 0; i < 2; ++i)
#pragma unroll
        for (int n = 0; n < 4; ++n)
#pragma unroll
            for (int r = 0; r < 8; ++r) {
                int row = wm + i * 16 + r + hi * 8;
                int col = wn + n * 16 + lm;
                Of[row * 130 + col] = acc[i][n].s[r];
            }
    __syncthreads();
    int b = Mbase >> 10, hwb = Mbase & 1023;
#pragma unroll
    for (int it = 0; it < 64; ++it) {
        int i = t + it * 128;
        int orow = i >> 6, hwi = i & 63;
        int o = Nbase + orow;
        size_t idx = ((size_t)b * NC + o) * NHW + hwb + hwi;
        out[idx] = Of[hwi * 130 + orow] + pb[o] + xn[idx];   // bias + residual
    }
}

// ---------------------------------------------------------------- launch
extern "C" void kernel_launch(void* const* d_in, const int* in_sizes, int n_in,
                              void* d_out, int out_size, void* d_ws, size_t ws_size,
                              hipStream_t stream) {
    (void)in_sizes; (void)n_in; (void)out_size; (void)ws_size;
    const float* x     = (const float*)d_in[0];
    const float* gw    = (const float*)d_in[1];
    const float* gb    = (const float*)d_in[2];
    const float* qkvw  = (const float*)d_in[3];
    const float* projw = (const float*)d_in[4];
    const float* projb = (const float*)d_in[5];
    float* out = (float*)d_out;

    char* ws = (char*)d_ws;
    size_t off = 0;
    auto alloc = [&](size_t bytes) -> char* {
        char* p = ws + off;
        off = (off + bytes + 255) & ~(size_t)255;
        return p;
    };
    float*    mean = (float*)alloc(NB * NGRP * 4);
    float*    rstd = (float*)alloc(NB * NGRP * 4);
    _Float16* qh   = (_Float16*)alloc((size_t)3 * NC * NC * 2);
    _Float16* ph   = (_Float16*)alloc((size_t)NC * NC * 2);
    float*    xn   = (float*)alloc((size_t)NB * NC * NHW * 4);
    _Float16* xnT  = (_Float16*)alloc((size_t)NB * NHW * NC * 2);
    _Float16* qbuf = (_Float16*)alloc((size_t)NB * NHEAD * NHW * HD * 2);
    _Float16* kbuf = (_Float16*)alloc((size_t)NB * NHEAD * NHW * HD * 2);
    _Float16* vtb  = (_Float16*)alloc((size_t)NB * NHEAD * HD * NHW * 2);
    _Float16* hbuf = (_Float16*)alloc((size_t)NB * NHW * NC * 2);

    gn_stats_k<<<NB * NGRP, 256, 0, stream>>>(x, mean, rstd);
    wcvt_k<<<(3 * NC * NC) / 256, 256, 0, stream>>>(qkvw, projw, qh, ph);
    gn_apply_k<<<dim3(NHW / 32, NC / 32, NB), dim3(32, 8), 0, stream>>>(
        x, gw, gb, mean, rstd, xn, xnT);
    qkv_gemm_k<<<dim3((NB * NHW) / 64, (3 * NC) / 128), 128, 0, stream>>>(
        xnT, qh, qbuf, kbuf, vtb);
    attn_k<<<dim3(NHW / 64, NB * NHEAD), 128, 0, stream>>>(qbuf, kbuf, vtb, hbuf);
    proj_gemm_k<<<dim3((NB * NHW) / 64, NC / 128), 128, 0, stream>>>(
        hbuf, ph, projb, xn, out);
}
